// GRUCell_44367012168293
// MI455X (gfx1250) — matrix operand order repaired and checked by reading, not compiled, not run on hardware
//
#include <hip/hip_runtime.h>
#include <hip/hip_bf16.h>

typedef __attribute__((ext_vector_type(16))) __bf16 v16bf;
typedef __attribute__((ext_vector_type(8)))  float  v8f;

#define BATCH      65536
#define MTILE      32          // batch rows per workgroup
#define LDS_STRIDE 260         // 256 + 4 pad -> conflict-free column access

// Split fp32 into bf16 hi (truncated, exact) + bf16 lo (residual).
// hi + lo reproduces x to ~16 mantissa bits; 3-term WMMA gives ~fp32 GEMM.
__device__ __forceinline__ void splitf(float x, __bf16 &hi, __bf16 &lo) {
  unsigned u  = __builtin_bit_cast(unsigned, x) & 0xffff0000u;
  float    hf = __builtin_bit_cast(float, u);
  hi = (__bf16)hf;          // exact (low mantissa bits are zero)
  lo = (__bf16)(x - hf);    // RNE on the residual
}

// ---------------------------------------------------------------------------
// Prep: swizzle 6 fp32 weight matrices (k-major, 256x256) into bf16 hi/lo
// B-fragment layout for v_wmma_f32_16x16x32_bf16:
//   tile t = n_tile*8 + k_tile (16 n-tiles x 8 k-tiles)
//   lane (0..31): column n = (lane&15); lanes>=16 hold K upper half
//   element e (0..15): k = k_tile*32 + (lane>>4)*16 + e
// Stored per (matrix m, part p): 128 tiles * 32 lanes * 32B  = 128KB.
// ws layout: part index pi = m*2+p, u32[pi*32768 + t*256 + lane*8 + j]
// ---------------------------------------------------------------------------
__global__ void gru_prep(const float* __restrict__ Wz, const float* __restrict__ Uz,
                         const float* __restrict__ Wr, const float* __restrict__ Ur,
                         const float* __restrict__ Wh, const float* __restrict__ Uh,
                         unsigned* __restrict__ ws) {
  unsigned tid = blockIdx.x * 256u + threadIdx.x;
  unsigned m   = tid >> 15;
  if (m >= 6u) return;
  const float* src = (m == 0) ? Wz : (m == 1) ? Uz : (m == 2) ? Wr
                   : (m == 3) ? Ur : (m == 4) ? Wh : Uh;
  unsigned t    = (tid >> 8) & 127u;
  unsigned lane = (tid >> 3) & 31u;
  unsigned j    = tid & 7u;               // u32 pair index -> elements 2j, 2j+1
  unsigned n    = (t >> 3) * 16u + (lane & 15u);
  unsigned k    = (t & 7u) * 32u + (lane >> 4) * 16u + 2u * j;

  float v0 = src[k * 256u + n];
  float v1 = src[(k + 1u) * 256u + n];
  __bf16 h0, l0, h1, l1;
  splitf(v0, h0, l0);
  splitf(v1, h1, l1);
  unsigned hw = ((unsigned)__builtin_bit_cast(unsigned short, h1) << 16)
              |  (unsigned)__builtin_bit_cast(unsigned short, h0);
  unsigned lw = ((unsigned)__builtin_bit_cast(unsigned short, l1) << 16)
              |  (unsigned)__builtin_bit_cast(unsigned short, l0);
  unsigned idx = t * 256u + lane * 8u + j;
  ws[(m * 2u + 0u) * 32768u + idx] = hw;
  ws[(m * 2u + 1u) * 32768u + idx] = lw;
}

// 3-term split-bf16 accumulation: C += Ahi*Bhi + Ahi*Blo + Alo*Bhi
__device__ __forceinline__ v8f mm3(v16bf ahi, v16bf alo,
                                   const v16bf* __restrict__ bhip,
                                   const v16bf* __restrict__ blop, v8f c) {
  v16bf bhi = *bhip;
  v16bf blo = *blop;
  c = __builtin_amdgcn_wmma_f32_16x16x32_bf16(false, ahi, false, bhi, (short)0, c, false, false);
  c = __builtin_amdgcn_wmma_f32_16x16x32_bf16(false, ahi, false, blo, (short)0, c, false, false);
  c = __builtin_amdgcn_wmma_f32_16x16x32_bf16(false, alo, false, bhi, (short)0, c, false, false);
  return c;
}

// Build A-fragment (16-bit A 16x32, contiguous k order used here) from a row of 256 floats:
// lane<16 : row=lane,    elements 0..15 -> k0+0..15
// lane>=16: row=lane-16, elements 0..15 -> k0+16..31
__device__ __forceinline__ void buildA(const float* base, int k0, int sel,
                                       v16bf &hi, v16bf &lo) {
  union { float4 v[4]; float f[16]; } u;
  const float4* p = (const float4*)(base + k0 + 16 * sel);
  u.v[0] = p[0]; u.v[1] = p[1];
  u.v[2] = p[2]; u.v[3] = p[3];
#pragma unroll
  for (int e = 0; e < 16; ++e) {
    __bf16 he, le;
    splitf(u.f[e], he, le);
    hi[e] = he;
    lo[e] = le;
  }
}

__device__ __forceinline__ float sigmoidf_fast(float x) {
  return 1.0f / (1.0f + __expf(-x));
}

// ---------------------------------------------------------------------------
// Main fused GRU kernel: one block = 32 batch rows, 8 wave32s.
// Wave (wm 0..1, wn 0..3) owns rows [16*wm,16*wm+16) x cols [64*wn,64*wn+64).
// Pass A: z,r,x@Wh accumulators (15 WMMA per (k,n) tile). r*h shared via LDS.
// Pass B: (r*h)@Uh on top of x@Wh accumulator, fused tanh/blend/store.
// ---------------------------------------------------------------------------
__global__ void __launch_bounds__(256)
gru_main(const float* __restrict__ x, const float* __restrict__ h,
         const float* __restrict__ bz, const float* __restrict__ br,
         const float* __restrict__ bh, const unsigned* __restrict__ ws,
         float* __restrict__ out) {
  __shared__ float rh[MTILE * LDS_STRIDE];

  const int l   = threadIdx.x & 31;
  const int w   = threadIdx.x >> 5;
  const int wm  = w >> 2;
  const int wn  = w & 3;
  const int sel = l >> 4;
  const int lc  = l & 15;
  const int m0  = blockIdx.x * MTILE + wm * 16;   // first global row of wave

  const v16bf* wsv = (const v16bf*)ws;            // 4096 v16bf per (matrix,part)

  v8f accz[4], accr[4], acch[4];
#pragma unroll
  for (int nt = 0; nt < 4; ++nt) {
    accz[nt] = (v8f){0.f,0.f,0.f,0.f,0.f,0.f,0.f,0.f};
    accr[nt] = (v8f){0.f,0.f,0.f,0.f,0.f,0.f,0.f,0.f};
    acch[nt] = (v8f){0.f,0.f,0.f,0.f,0.f,0.f,0.f,0.f};
  }

  const float* xrow = x + (size_t)(m0 + lc) * 256;
  const float* hrow = h + (size_t)(m0 + lc) * 256;

  // ---- Pass A: z_pre = x@Wz + h@Uz ; r_pre = x@Wr + h@Ur ; hx = x@Wh ----
  for (int kt = 0; kt < 8; ++kt) {
    const int k0 = kt * 32;
    if (kt < 7) {                     // hint: stream next k-tile of A operands
      __builtin_prefetch(xrow + k0 + 32, 0, 0);
      __builtin_prefetch(hrow + k0 + 32, 0, 0);
    }
    v16bf xhi, xlo, hhi, hlo;
    buildA(xrow, k0, sel, xhi, xlo);
    buildA(hrow, k0, sel, hhi, hlo);
#pragma unroll
    for (int nt = 0; nt < 4; ++nt) {
      const int t = (wn * 4 + nt) * 8 + kt;
      const v16bf* W = wsv + t * 32 + l;
      accz[nt] = mm3(xhi, xlo, W + 0 * 4096, W + 1 * 4096, accz[nt]);  // Wz
      accz[nt] = mm3(hhi, hlo, W + 2 * 4096, W + 3 * 4096, accz[nt]);  // Uz
      accr[nt] = mm3(xhi, xlo, W + 4 * 4096, W + 5 * 4096, accr[nt]);  // Wr
      accr[nt] = mm3(hhi, hlo, W + 6 * 4096, W + 7 * 4096, accr[nt]);  // Ur
      acch[nt] = mm3(xhi, xlo, W + 8 * 4096, W + 9 * 4096, acch[nt]);  // Wh
    }
  }

  // ---- Epilogue A: z = sigmoid, r = sigmoid, stash r*h in LDS ----
#pragma unroll
  for (int nt = 0; nt < 4; ++nt) {
    const int n = wn * 64 + nt * 16 + lc;
    const float bzv = bz[n];
    const float brv = br[n];
#pragma unroll
    for (int j = 0; j < 8; ++j) {
      const int mrow = sel * 8 + j;                     // 0..15 within wave tile
      const float hval = h[(size_t)(m0 + mrow) * 256 + n];
      const float zv = sigmoidf_fast(accz[nt][j] + bzv);
      const float rv = sigmoidf_fast(accr[nt][j] + brv);
      accz[nt][j] = zv;                                 // keep z for the blend
      rh[(wm * 16 + mrow) * LDS_STRIDE + n] = rv * hval;
    }
  }
  __syncthreads();

  // ---- Pass B: acch += (r*h) @ Uh ----
  const float* rhrow = rh + (wm * 16 + lc) * LDS_STRIDE;
  for (int kt = 0; kt < 8; ++kt) {
    const int k0 = kt * 32;
    v16bf ahi, alo;
    buildA(rhrow, k0, sel, ahi, alo);
#pragma unroll
    for (int nt = 0; nt < 4; ++nt) {
      const int t = (wn * 4 + nt) * 8 + kt;
      const v16bf* W = wsv + t * 32 + l;
      acch[nt] = mm3(ahi, alo, W + 10 * 4096, W + 11 * 4096, acch[nt]); // Uh
    }
  }

  // ---- Epilogue B: h_hat = tanh(.+bh); h_t = z*h + (1-z)*h_hat; dup store --
#pragma unroll
  for (int nt = 0; nt < 4; ++nt) {
    const int n = wn * 64 + nt * 16 + lc;
    const float bhv = bh[n];
#pragma unroll
    for (int j = 0; j < 8; ++j) {
      const int grow = m0 + sel * 8 + j;
      const float hval = h[(size_t)grow * 256 + n];
      const float hhat = tanhf(acch[nt][j] + bhv);
      const float zv   = accz[nt][j];
      const float ht   = zv * hval + (1.0f - zv) * hhat;
      out[(size_t)grow * 256 + n] = ht;
      out[(size_t)BATCH * 256 + (size_t)grow * 256 + n] = ht;   // tuple copy
    }
  }
}

extern "C" void kernel_launch(void* const* d_in, const int* in_sizes, int n_in,
                              void* d_out, int out_size, void* d_ws, size_t ws_size,
                              hipStream_t stream) {
  const float* x  = (const float*)d_in[0];
  const float* h  = (const float*)d_in[1];
  const float* Wz = (const float*)d_in[2];
  const float* Uz = (const float*)d_in[3];
  const float* bz = (const float*)d_in[4];
  const float* Wr = (const float*)d_in[5];
  const float* Ur = (const float*)d_in[6];
  const float* br = (const float*)d_in[7];
  const float* Wh = (const float*)d_in[8];
  const float* Uh = (const float*)d_in[9];
  const float* bh = (const float*)d_in[10];
  unsigned* ws = (unsigned*)d_ws;        // needs 6*2*128KB = 1.5 MB scratch
  float* out = (float*)d_out;

  // 6 matrices * 128 tiles * 32 lanes * 8 pairs = 196608 threads
  gru_prep<<<768, 256, 0, stream>>>(Wz, Uz, Wr, Ur, Wh, Uh, ws);
  gru_main<<<BATCH / MTILE, 256, 0, stream>>>(x, h, bz, br, bh, ws, out);
}
